// GotenNet_90426241450253
// MI455X (gfx1250) — matrix operand order repaired and checked by reading, not compiled, not run on hardware
//
#include <hip/hip_runtime.h>
#include <hip/hip_bf16.h>
#include <math.h>

// ---------------- problem constants ----------------
constexpr int B_   = 2;
constexpr int N_   = 2048;
constexpr int M_   = 32;     // neighbors
constexpr int D_   = 128;
constexpr int H_   = 8;
constexpr int DH_  = 16;
constexpr int DI_  = 128;    // H*DH
constexpr int S_   = 5;
constexpr int SDI_ = 640;    // S*DI
constexpr int DMLP_= 256;
constexpr int HS_  = 40;     // H*S
constexpr int BN_  = B_ * N_;

typedef __attribute__((ext_vector_type(16))) _Float16 v16h;
typedef __attribute__((ext_vector_type(8)))  float    v8f;
typedef unsigned int u32x4 __attribute__((ext_vector_type(4)));
typedef int          i32x8 __attribute__((ext_vector_type(8)));
typedef int          i32x4 __attribute__((ext_vector_type(4)));

// ---------------- fast-math helpers (v_rcp_f32 based, no IEEE div chains) ----
__device__ inline float frcp(float x)      { return __builtin_amdgcn_rcpf(x); }
__device__ inline float siluf(float x)     { return x * frcp(1.0f + __expf(-x)); }
__device__ inline float sigmoidf_(float x) { return frcp(1.0f + __expf(-x)); }
// NaN-safe fast tanh: 1 - 2/(1+e^{2x}) -> +/-1 at extremes (rcp(inf)=0)
__device__ inline float tanh_fast(float x) {
  return 1.0f - 2.0f * frcp(1.0f + __expf(2.0f * x));
}
__device__ inline float softclampf(float x) { return 50.0f * tanh_fast(x * 0.02f); }

__device__ inline float wave_max(float x) {
#pragma unroll
  for (int m = 16; m > 0; m >>= 1) x = fmaxf(x, __shfl_xor(x, m, 32));
  return x;
}
__device__ inline float wave_sum(float x) {
#pragma unroll
  for (int m = 16; m > 0; m >>= 1) x += __shfl_xor(x, m, 32);
  return x;
}

// ---------------- TDM: 1-D contiguous f32 tile -> LDS ----------------
// Builds D# group0/group1 per CDNA5 ISA and issues
// tensor_load_to_lds (6-arg clang-23 builtin). Call from ONE wave.
__device__ inline void tdm_load_1d_f32(const void* gptr, void* ldsptr, int nElems) {
  const unsigned long long ga = (unsigned long long)gptr;
  // generic LDS pointers carry the wave-relative LDS byte offset in the low 32 bits
  const unsigned lds = (unsigned)(unsigned long long)ldsptr;
  u32x4 g0;
  g0[0] = 1u;                                   // count=1 (valid user descriptor)
  g0[1] = lds;                                  // lds_addr (bytes)
  g0[2] = (unsigned)(ga & 0xFFFFFFFFull);       // global_addr[31:0]
  g0[3] = (unsigned)((ga >> 32) & 0x1FFFFFFull) // global_addr[56:32]
          | 0x80000000u;                        // type=2 ("image") at bits 127:126
  i32x8 g1;
  g1[0] = (int)(2u << 16);                      // data_size=2 -> 4 bytes/elem
  g1[1] = (int)(((unsigned)nElems & 0xFFFFu) << 16);   // tensor_dim0[15:0] @ bits 63:48
  g1[2] = (int)((((unsigned)nElems >> 16) & 0xFFFFu)   // tensor_dim0[31:16] @ bits 79:64
          | (1u << 16));                               // tensor_dim1 = 1
  g1[3] = (int)(((unsigned)nElems & 0xFFFFu) << 16);   // tile_dim0 @ bits 127:112
  g1[4] = 0;                                    // tile_dim1=0 (unused), tile_dim2=0
  g1[5] = nElems;                               // tensor_dim0_stride[31:0] @ bits 191:160
  g1[6] = 0;
  g1[7] = 0;
  i32x4 z4 = {0, 0, 0, 0};
  i32x8 z8 = {0, 0, 0, 0, 0, 0, 0, 0};
  __builtin_amdgcn_tensor_load_to_lds(g0, g1, z4, z4, z8, 0);
}

// ---------------- WMMA fragment loaders ----------------
// A-fragment (16x32 f16, M x K) from LDS, row-major [m][ld].
__device__ inline v16h load_a_frag(const _Float16* sA, int ld, int mBase, int kBase, int lane) {
  const int mrow = mBase + (lane & 15);
  const int khi  = (lane >> 4) << 3;  // +8 for upper half-wave
  v16h a;
#pragma unroll
  for (int r = 0; r < 8; ++r) {
    const int k = kBase + khi + 2 * r + ((r >= 4) ? 8 : 0);
    a[2 * r]     = sA[mrow * ld + k];
    a[2 * r + 1] = sA[mrow * ld + k + 1];
  }
  return a;
}

// B-fragment (32x16 f16, K x N) from global f16 weight stored transposed:
// WT[n][k], k contiguous (ld = 128). Lanes 0-15: K=kBase+0..7,16..23; lanes 16-31: +8.
__device__ inline v16h load_b_frag(const _Float16* WT, int nBase, int kBase, int lane) {
  const int col = nBase + (lane & 15);
  const int k0  = kBase + ((lane >> 4) << 3);
  const _Float16* p = WT + (size_t)col * D_ + k0;
  v16h b;
#pragma unroll
  for (int q = 0; q < 16; ++q) b[q] = p[q + ((q >= 8) ? 8 : 0)];
  return b;
}

// 16x16 tile, K = 128 in four 16x16x32 WMMA steps.
__device__ inline v8f wmma_k128(const _Float16* sA, const _Float16* WT,
                                int mBase, int nCol, int lane) {
  v8f acc = {};
#pragma unroll
  for (int kt = 0; kt < 4; ++kt) {
    v16h a = load_a_frag(sA, D_, mBase, kt * 32, lane);
    v16h b = load_b_frag(WT, nCol, kt * 32, lane);
    acc = __builtin_amdgcn_wmma_f32_16x16x32_f16(
        false, a, false, b, (short)0, acc, false, false);
  }
  return acc;
}

// ---------------- kernel 0: weight transpose + f32->f16 ----------------
__global__ void goten_convert_kernel(const float* __restrict__ Wek,
                                     const float* __restrict__ Wev,
                                     const float* __restrict__ Wc,
                                     _Float16* __restrict__ wekT,
                                     _Float16* __restrict__ wevT,
                                     _Float16* __restrict__ wcT) {
  const int idx = blockIdx.x * blockDim.x + threadIdx.x;
  if (idx < SDI_ * D_) {
    const int n = idx / D_, k = idx % D_;
    wekT[idx] = (_Float16)Wek[(size_t)k * SDI_ + n];
    wevT[idx] = (_Float16)Wev[(size_t)k * SDI_ + n];
  }
  if (idx < D_ * D_) {
    const int n = idx / D_, k = idx % D_;
    wcT[idx] = (_Float16)Wc[(size_t)k * D_ + n];
  }
}

// ---------------- kernel 1: per-node LN + projections ----------------
__global__ void goten_node_kernel(const float* __restrict__ h,
                                  const float* __restrict__ g_hi,
                                  const float* __restrict__ g_hj,
                                  const float* __restrict__ Wq,
                                  const float* __restrict__ Wk,
                                  const float* __restrict__ Wv1, const float* __restrict__ bv1,
                                  const float* __restrict__ Wv2, const float* __restrict__ bv2,
                                  const float* __restrict__ Wp1, const float* __restrict__ bp1,
                                  const float* __restrict__ Wp2, const float* __restrict__ bp2,
                                  const float* __restrict__ Wg,  const float* __restrict__ bg,
                                  float* __restrict__ qo, float* __restrict__ ko,
                                  float* __restrict__ vo, float* __restrict__ po,
                                  float* __restrict__ go) {
  const int node = blockIdx.x;  // b*N + n
  const int t = threadIdx.x;
  __shared__ float sh[D_], shi[D_], shj[D_], stmp[DMLP_];
  __shared__ float sMu, sRs;

  if (t < D_) sh[t] = h[(size_t)node * D_ + t];
  __syncthreads();
  if (t == 0) {
    float s = 0.f;
    for (int d = 0; d < D_; ++d) s += sh[d];
    const float mu = s / D_;
    float v = 0.f;
    for (int d = 0; d < D_; ++d) { const float x = sh[d] - mu; v += x * x; }
    sMu = mu;
    sRs = rsqrtf(v / D_ + 1e-5f);
  }
  __syncthreads();
  if (t < D_) {
    const float xn = (sh[t] - sMu) * sRs;
    shi[t] = xn * g_hi[t];
    shj[t] = xn * g_hj[t];
  }
  __syncthreads();

  if (t < D_) {
    float aq = 0.f, ak = 0.f;
    for (int kk = 0; kk < D_; ++kk) {
      aq += shi[kk] * Wq[kk * DI_ + t];
      ak += shj[kk] * Wk[kk * DI_ + t];
    }
    qo[(size_t)node * DI_ + t] = aq;
    ko[(size_t)node * DI_ + t] = ak;
  }
  if (t < HS_) {
    float a = bg[t];
    for (int kk = 0; kk < D_; ++kk) a += shi[kk] * Wg[kk * HS_ + t];
    go[(size_t)node * HS_ + t] = sigmoidf_(a);
  }
  {
    float a = bv1[t];
    for (int kk = 0; kk < D_; ++kk) a += shj[kk] * Wv1[kk * DMLP_ + t];
    stmp[t] = siluf(a);
  }
  __syncthreads();
  for (int o = t; o < SDI_; o += 256) {
    float a = bv2[o];
    for (int kk = 0; kk < DMLP_; ++kk) a += stmp[kk] * Wv2[kk * SDI_ + o];
    vo[(size_t)node * SDI_ + o] = a;
  }
  __syncthreads();
  {
    float a = bp1[t];
    for (int kk = 0; kk < D_; ++kk) a += shj[kk] * Wp1[kk * DMLP_ + t];
    stmp[t] = siluf(a);
  }
  __syncthreads();
  for (int o = t; o < SDI_; o += 256) {
    float a = bp2[o];
    for (int kk = 0; kk < DMLP_; ++kk) a += stmp[kk] * Wp2[kk * SDI_ + o];
    po[(size_t)node * SDI_ + o] = a;
  }
}

// ---------------- kernel 2: fused edge GEMMs + attention + output ----------------
// One block per (b, i). 256 threads = 8 wave32 waves.
__global__ void goten_edge_kernel(const float* __restrict__ t_ij,
                                  const float* __restrict__ r1,
                                  const float* __restrict__ r2,
                                  const float* __restrict__ x1,
                                  const float* __restrict__ x2,
                                  const int* __restrict__ nidx,
                                  const unsigned char* __restrict__ nmask,
                                  const float* __restrict__ q_ws,
                                  const float* __restrict__ k_ws,
                                  const float* __restrict__ v_ws,
                                  const float* __restrict__ p_ws,
                                  const float* __restrict__ g_ws,
                                  const _Float16* __restrict__ wekT,
                                  const _Float16* __restrict__ wevT,
                                  const _Float16* __restrict__ wcT,
                                  float* __restrict__ out_h,
                                  float* __restrict__ out_x1,
                                  float* __restrict__ out_x2) {
  const int node = blockIdx.x;          // b*N + i
  const int bN   = (node / N_) * N_;    // batch base for gathers
  const int t    = threadIdx.x;
  const int lane = t & 31;
  const int wave = t >> 5;

  __shared__ _Float16 sT[M_ * D_];     // t_ij tile (f16)
  __shared__ _Float16 sKg[M_ * D_];    // gathered k (f16)
  __shared__ _Float16 sSea[M_ * D_];   // reused: ek_s (f16) then sea_s (f16)
  __shared__ float    sOut[M_ * D_];   // reused: TDM f32 stage, ev_s, out_s
  __shared__ float    sQ[D_];
  __shared__ float    sGate[HS_];
  __shared__ float    sAttn[M_ * HS_]; // attn[j][h][s]
  __shared__ float    sR1[M_ * 3];
  __shared__ float    sR2[M_ * 5];
  __shared__ int      sIdx[M_];

  // hint: keep the shared transposed weight tables hot (global_prefetch_b8)
  __builtin_prefetch(wekT, 0, 3);
  __builtin_prefetch(wevT, 0, 3);
  __builtin_prefetch(wcT, 0, 3);

  // ---- async: TDM the contiguous 32x128 f32 t_ij tile into LDS (sOut) ----
  if (wave == 0) {
    tdm_load_1d_f32(t_ij + (size_t)node * M_ * D_, sOut, M_ * D_);
  }

  // ---- stage small per-node data (overlaps TDM) ----
  if (t < M_)  sIdx[t]  = nidx[(size_t)node * M_ + t];
  if (t < HS_) sGate[t] = g_ws[(size_t)node * HS_ + t];
  if (t < D_)  sQ[t]    = q_ws[(size_t)node * DI_ + t];
  if (t < M_ * 3) sR1[t] = r1[(size_t)node * M_ * 3 + t];
  if (t < M_ * 5) sR2[t] = r2[(size_t)node * M_ * 5 + t];

  if (wave == 0) __builtin_amdgcn_s_wait_tensorcnt(0);
  __syncthreads();

  // ---- convert tile to f16, gather k (f16) ----
  for (int e = t; e < M_ * D_; e += 256) {
    const int j = e >> 7;
    sT[e]  = (_Float16)sOut[e];
    sKg[e] = (_Float16)k_ws[((size_t)(bN + sIdx[j])) * DI_ + (e & 127)];
  }
  __syncthreads();

  // ---- phase 1: sim[h][j][s] via per-s ek WMMA ----
  float simReg[S_];
  const int hh = wave;      // head handled by this wave
  const int jj = lane;      // neighbor handled by this lane
  const bool ok = nmask[(size_t)node * M_ + jj] != 0;

  for (int s = 0; s < S_; ++s) {
    for (int job = wave; job < 16; job += 8) {
      const int mt = job >> 3, nt = job & 7;
      v8f acc = wmma_k128(sT, wekT, mt * 16, s * 128 + nt * 16, lane);
      const int ncol  = nt * 16 + (lane & 15);
      const int mbase = mt * 16 + ((lane >> 4) << 3);
#pragma unroll
      for (int r = 0; r < 8; ++r)
        sSea[(mbase + r) * D_ + ncol] = (_Float16)siluf(acc[r]);
    }
    __syncthreads();
    float acc = 0.f;
#pragma unroll
    for (int dh = 0; dh < DH_; ++dh) {
      const int c = hh * DH_ + dh;
      acc += sQ[c] * (float)sKg[jj * D_ + c] * (float)sSea[jj * D_ + c];
    }
    simReg[s] = ok ? softclampf(acc) : -3.4e38f;
    __syncthreads();
  }

  // ---- phase 2: softmax over j (lanes of each wave; wave32 == M) ----
#pragma unroll
  for (int s = 0; s < S_; ++s) {
    const float m  = wave_max(simReg[s]);
    const float e  = __expf(simReg[s] - m);
    const float sm = wave_sum(e);
    sAttn[jj * HS_ + hh * S_ + s] = e * frcp(sm);
  }
  __syncthreads();

  // ---- phase 3: per-s ev WMMA, sea, out WMMA, reductions ----
  float hacc = 0.f;
  float xr1acc[2] = {0.f, 0.f};
  float xr2acc[3] = {0.f, 0.f, 0.f};

  for (int s = 0; s < S_; ++s) {
    for (int job = wave; job < 16; job += 8) {
      const int mt = job >> 3, nt = job & 7;
      v8f acc = wmma_k128(sT, wevT, mt * 16, s * 128 + nt * 16, lane);
      const int ncol  = nt * 16 + (lane & 15);
      const int mbase = mt * 16 + ((lane >> 4) << 3);
#pragma unroll
      for (int r = 0; r < 8; ++r) sOut[(mbase + r) * D_ + ncol] = acc[r];
    }
    __syncthreads();

    for (int e = t; e < M_ * D_; e += 256) {
      const int j = e >> 7, c = e & 127;
      const int hI  = c >> 4;
      const size_t gnode = (size_t)(bN + sIdx[j]);
      const float vg  = v_ws[gnode * SDI_ + s * DI_ + c];
      const float pg  = p_ws[gnode * SDI_ + s * DI_ + c];
      const float at  = sAttn[j * HS_ + hI * S_ + s];
      const float gt  = sGate[hI * S_ + s];
      sSea[e] = (_Float16)((at * vg + sOut[e] * pg) * gt);
    }
    __syncthreads();

    for (int job = wave; job < 16; job += 8) {
      const int mt = job >> 3, nt = job & 7;
      v8f acc = wmma_k128(sSea, wcT, mt * 16, nt * 16, lane);
      const int ncol  = nt * 16 + (lane & 15);
      const int mbase = mt * 16 + ((lane >> 4) << 3);
#pragma unroll
      for (int r = 0; r < 8; ++r) sOut[(mbase + r) * D_ + ncol] = acc[r];
    }
    __syncthreads();

    if (s == 0) {
      if (t < D_) {
        float a = 0.f;
        for (int j = 0; j < M_; ++j) a += sOut[j * D_ + t];
        hacc += a;
      }
    } else if (s == 1 || s == 3) {
#pragma unroll
      for (int ii = 0; ii < 2; ++ii) {
        const int idx = t + ii * 256;
        if (idx < D_ * 3) {
          const int d = idx / 3, m3 = idx % 3;
          float a = 0.f;
          if (s == 1) {
            for (int j = 0; j < M_; ++j) a += sR1[j * 3 + m3] * sOut[j * D_ + d];
          } else {
            for (int j = 0; j < M_; ++j)
              a += x1[((size_t)(bN + sIdx[j]) * D_ + d) * 3 + m3] * sOut[j * D_ + d];
          }
          xr1acc[ii] += a;
        }
      }
    } else {  // s == 2 || s == 4
#pragma unroll
      for (int ii = 0; ii < 3; ++ii) {
        const int idx = t + ii * 256;
        if (idx < D_ * 5) {
          const int d = idx / 5, m5 = idx % 5;
          float a = 0.f;
          if (s == 2) {
            for (int j = 0; j < M_; ++j) a += sR2[j * 5 + m5] * sOut[j * D_ + d];
          } else {
            for (int j = 0; j < M_; ++j)
              a += x2[((size_t)(bN + sIdx[j]) * D_ + d) * 5 + m5] * sOut[j * D_ + d];
          }
          xr2acc[ii] += a;
        }
      }
    }
    __syncthreads();
  }

  // ---- final writes ----
  if (t < D_) out_h[(size_t)node * D_ + t] = hacc;
#pragma unroll
  for (int ii = 0; ii < 2; ++ii) {
    const int idx = t + ii * 256;
    if (idx < D_ * 3) {
      const int d = idx / 3, m3 = idx % 3;
      out_x1[((size_t)node * D_ + d) * 3 + m3] = xr1acc[ii];
    }
  }
#pragma unroll
  for (int ii = 0; ii < 3; ++ii) {
    const int idx = t + ii * 256;
    if (idx < D_ * 5) {
      const int d = idx / 5, m5 = idx % 5;
      out_x2[((size_t)node * D_ + d) * 5 + m5] = xr2acc[ii];
    }
  }
}

// ---------------- host launcher ----------------
extern "C" void kernel_launch(void* const* d_in, const int* in_sizes, int n_in,
                              void* d_out, int out_size, void* d_ws, size_t ws_size,
                              hipStream_t stream) {
  const float* h    = (const float*)d_in[0];
  const float* t_ij = (const float*)d_in[1];
  const float* r1   = (const float*)d_in[2];
  const float* r2   = (const float*)d_in[3];
  const float* x1   = (const float*)d_in[4];
  const float* x2   = (const float*)d_in[5];
  const float* g_hi = (const float*)d_in[6];
  const float* g_hj = (const float*)d_in[7];
  const float* Wq   = (const float*)d_in[8];
  const float* Wk   = (const float*)d_in[9];
  const float* Wv1  = (const float*)d_in[10];
  const float* bv1  = (const float*)d_in[11];
  const float* Wv2  = (const float*)d_in[12];
  const float* bv2  = (const float*)d_in[13];
  const float* Wp1  = (const float*)d_in[14];
  const float* bp1  = (const float*)d_in[15];
  const float* Wp2  = (const float*)d_in[16];
  const float* bp2  = (const float*)d_in[17];
  const float* Wek  = (const float*)d_in[18];
  const float* Wev  = (const float*)d_in[19];
  const float* Wg   = (const float*)d_in[20];
  const float* bg   = (const float*)d_in[21];
  const float* Wc   = (const float*)d_in[22];
  const int*   nidx = (const int*)d_in[23];
  const unsigned char* nmask = (const unsigned char*)d_in[24];

  float* q_ws = (float*)d_ws;                 // BN * 128
  float* k_ws = q_ws + (size_t)BN_ * DI_;     // BN * 128
  float* v_ws = k_ws + (size_t)BN_ * DI_;     // BN * 640
  float* p_ws = v_ws + (size_t)BN_ * SDI_;    // BN * 640
  float* g_ws = p_ws + (size_t)BN_ * SDI_;    // BN * 40
  _Float16* wekT = (_Float16*)(g_ws + (size_t)BN_ * HS_);  // 640*128
  _Float16* wevT = wekT + (size_t)SDI_ * D_;               // 640*128
  _Float16* wcT  = wevT + (size_t)SDI_ * D_;               // 128*128

  float* out_h  = (float*)d_out;                       // [B,N,128]
  float* out_x1 = out_h + (size_t)BN_ * D_;            // [B,N,128,3]
  float* out_x2 = out_x1 + (size_t)BN_ * D_ * 3;       // [B,N,128,5]

  const int convN = SDI_ * D_;
  goten_convert_kernel<<<(convN + 255) / 256, 256, 0, stream>>>(
      Wek, Wev, Wc, wekT, wevT, wcT);

  goten_node_kernel<<<BN_, 256, 0, stream>>>(
      h, g_hi, g_hj, Wq, Wk, Wv1, bv1, Wv2, bv2,
      Wp1, bp1, Wp2, bp2, Wg, bg, q_ws, k_ws, v_ws, p_ws, g_ws);

  goten_edge_kernel<<<BN_, 256, 0, stream>>>(
      t_ij, r1, r2, x1, x2, nidx, nmask,
      q_ws, k_ws, v_ws, p_ws, g_ws, wekT, wevT, wcT,
      out_h, out_x1, out_x2);
}
